// GATnet_17995912970423
// MI455X (gfx1250) — matrix operand order repaired and checked by reading, not compiled, not run on hardware
//
#include <hip/hip_runtime.h>

typedef __attribute__((ext_vector_type(16))) _Float16 v16h;
typedef __attribute__((ext_vector_type(8)))  float    v8f;

#define NEG_SLOPE 0.2f

// ---- monotonic float<->uint encoding for atomicMax on signed floats ----
__device__ __forceinline__ unsigned f2ord(float f) {
    unsigned u = __float_as_uint(f);
    return (u & 0x80000000u) ? ~u : (u | 0x80000000u);
}
__device__ __forceinline__ float ord2f(unsigned u) {
    return (u & 0x80000000u) ? __uint_as_float(u & 0x7FFFFFFFu)
                             : __uint_as_float(~u);
}

// =====================  GEMM 1: h1 = X[N,128] @ W1[128,64]  =====================
// one wave -> 16x64 output tile via v_wmma_f32_16x16x32_f16
__global__ void gat_gemm1_wmma(const float* __restrict__ X,
                               const float* __restrict__ W,
                               float* __restrict__ H, int Nn) {
    int gwave = (blockIdx.x * blockDim.x + threadIdx.x) >> 5;
    int lane  = threadIdx.x & 31;
    int m0 = gwave * 16;
    if (m0 >= Nn) return;
    int half = lane >> 4;     // K-half selector
    int l15  = lane & 15;     // A: row M ; B/C: col N
    int rowA = m0 + l15;

    v8f acc0 = {}, acc1 = {}, acc2 = {}, acc3 = {};
    for (int kc = 0; kc < 4; ++kc) {
        int kbase = kc * 32;
        // A fragment: lane holds M=l15, K in {kbase+half*8+0..7, +16..}
        v16h a;
        const float* xp = X + (size_t)rowA * 128 + kbase + half * 8;
#pragma unroll
        for (int g = 0; g < 2; ++g)
#pragma unroll
            for (int w = 0; w < 8; ++w)
                a[g * 8 + w] = (_Float16)xp[g * 16 + w];

#pragma unroll
        for (int nt = 0; nt < 4; ++nt) {
            // B fragment: lane holds N=nt*16+l15, K in {kbase+half*8+0..7, +16..}
            v16h b;
            int ncol = nt * 16 + l15;
            const float* wp = W + (size_t)(kbase + half * 8) * 64 + ncol;
#pragma unroll
            for (int i = 0; i < 16; ++i) b[i] = (_Float16)wp[(i + (i >> 3) * 8) * 64];
            v8f c = (nt == 0) ? acc0 : (nt == 1) ? acc1 : (nt == 2) ? acc2 : acc3;
            c = __builtin_amdgcn_wmma_f32_16x16x32_f16(false, a, false, b,
                                                       (short)0, c, false, false);
            if (nt == 0) acc0 = c; else if (nt == 1) acc1 = c;
            else if (nt == 2) acc2 = c; else acc3 = c;
        }
    }
    // C/D layout: vgpr r -> M = r + 8*half ; lane l15 -> N
#pragma unroll
    for (int r = 0; r < 8; ++r) {
        int m = m0 + r + half * 8;
        H[(size_t)m * 64 +  0 + l15] = acc0[r];
        H[(size_t)m * 64 + 16 + l15] = acc1[r];
        H[(size_t)m * 64 + 32 + l15] = acc2[r];
        H[(size_t)m * 64 + 48 + l15] = acc3[r];
    }
}

// =====================  GEMM 2: h2 = X2[N,64] @ W2[64,40]  =====================
__global__ void gat_gemm2_wmma(const float* __restrict__ X,
                               const float* __restrict__ W,
                               float* __restrict__ H, int Nn) {
    int gwave = (blockIdx.x * blockDim.x + threadIdx.x) >> 5;
    int lane  = threadIdx.x & 31;
    int m0 = gwave * 16;
    if (m0 >= Nn) return;
    int half = lane >> 4;
    int l15  = lane & 15;
    int rowA = m0 + l15;

    v8f acc0 = {}, acc1 = {}, acc2 = {};
    for (int kc = 0; kc < 2; ++kc) {
        int kbase = kc * 32;
        v16h a;
        const float* xp = X + (size_t)rowA * 64 + kbase + half * 8;
#pragma unroll
        for (int g = 0; g < 2; ++g)
#pragma unroll
            for (int w = 0; w < 8; ++w)
                a[g * 8 + w] = (_Float16)xp[g * 16 + w];

#pragma unroll
        for (int nt = 0; nt < 3; ++nt) {
            v16h b;
            int ncol = nt * 16 + l15;
            bool nok = ncol < 40;
            const float* wp = W + (size_t)(kbase + half * 8) * 40 + (nok ? ncol : 0);
#pragma unroll
            for (int i = 0; i < 16; ++i)
                b[i] = nok ? (_Float16)wp[(i + (i >> 3) * 8) * 40] : (_Float16)0.0f;
            v8f c = (nt == 0) ? acc0 : (nt == 1) ? acc1 : acc2;
            c = __builtin_amdgcn_wmma_f32_16x16x32_f16(false, a, false, b,
                                                       (short)0, c, false, false);
            if (nt == 0) acc0 = c; else if (nt == 1) acc1 = c; else acc2 = c;
        }
    }
#pragma unroll
    for (int r = 0; r < 8; ++r) {
        int m = m0 + r + half * 8;
        H[(size_t)m * 40 +  0 + l15] = acc0[r];
        H[(size_t)m * 40 + 16 + l15] = acc1[r];
        int nc = 32 + l15;
        if (nc < 40) H[(size_t)m * 40 + nc] = acc2[r];
    }
}

// =====================  attention logits + inits  =====================
__global__ void gat_att1(const float* __restrict__ h1,
                         const float* __restrict__ as, const float* __restrict__ ad,
                         float* __restrict__ osrc, float* __restrict__ odst, int Nn) {
    int n = blockIdx.x * blockDim.x + threadIdx.x;
    if (n >= Nn) return;
#pragma unroll
    for (int h = 0; h < 8; ++h) {
        float s1 = 0.f, s2 = 0.f;
#pragma unroll
        for (int c = 0; c < 8; ++c) {
            float v = h1[(size_t)n * 64 + h * 8 + c];
            s1 += v * as[h * 8 + c];
            s2 += v * ad[h * 8 + c];
        }
        osrc[(size_t)n * 8 + h] = s1;
        odst[(size_t)n * 8 + h] = s2;
    }
}

__global__ void gat_att2(const float* __restrict__ h2,
                         const float* __restrict__ as, const float* __restrict__ ad,
                         float* __restrict__ osrc, float* __restrict__ odst, int Nn) {
    int n = blockIdx.x * blockDim.x + threadIdx.x;
    if (n >= Nn) return;
    float s1 = 0.f, s2 = 0.f;
#pragma unroll
    for (int c = 0; c < 40; ++c) {
        float v = h2[(size_t)n * 40 + c];
        s1 += v * as[c];
        s2 += v * ad[c];
    }
    osrc[n] = s1;
    odst[n] = s2;
}

__global__ void gat_init1(float* __restrict__ out1, unsigned* __restrict__ emax,
                          float* __restrict__ denom, int Nn) {
    int i = blockIdx.x * blockDim.x + threadIdx.x;
    size_t tot = (size_t)Nn * 64;
    if ((size_t)i >= tot) return;
    out1[i] = 0.f;
    if (i < Nn * 8) { emax[i] = 0u; denom[i] = 0.f; }
}

__global__ void gat_init2(float* __restrict__ out2, unsigned* __restrict__ emax,
                          float* __restrict__ denom, int Nn) {
    int i = blockIdx.x * blockDim.x + threadIdx.x;
    size_t tot = (size_t)Nn * 40;
    if ((size_t)i >= tot) return;
    out2[i] = 0.f;
    if (i < Nn) { emax[i] = 0u; denom[i] = 0.f; }
}

// =====================  layer-1 edge passes (H=8)  =====================
__device__ __forceinline__ void load_edge(const long long* ei, int e, int E, int Nn,
                                          int& s, int& d) {
    if (e < E) { s = (int)ei[e]; d = (int)ei[(size_t)E + e]; }
    else       { s = d = e - E; }          // appended self-loops
}

__global__ void gat_edge_max1(const long long* __restrict__ ei, int E, int Nn,
                              const float* __restrict__ asrc, const float* __restrict__ adst,
                              unsigned* __restrict__ emax) {
    int e = blockIdx.x * blockDim.x + threadIdx.x;
    if (e >= E + Nn) return;
    int s, d; load_edge(ei, e, E, Nn, s, d);
#pragma unroll
    for (int h = 0; h < 8; ++h) {
        float v = asrc[(size_t)s * 8 + h] + adst[(size_t)d * 8 + h];
        v = v >= 0.f ? v : NEG_SLOPE * v;
        atomicMax(&emax[(size_t)d * 8 + h], f2ord(v));
    }
}

__global__ void gat_edge_sum1(const long long* __restrict__ ei, int E, int Nn,
                              const float* __restrict__ asrc, const float* __restrict__ adst,
                              const unsigned* __restrict__ emax, float* __restrict__ denom) {
    int e = blockIdx.x * blockDim.x + threadIdx.x;
    if (e >= E + Nn) return;
    int s, d; load_edge(ei, e, E, Nn, s, d);
#pragma unroll
    for (int h = 0; h < 8; ++h) {
        float v = asrc[(size_t)s * 8 + h] + adst[(size_t)d * 8 + h];
        v = v >= 0.f ? v : NEG_SLOPE * v;
        float ex = __expf(v - ord2f(emax[(size_t)d * 8 + h]));
        atomicAdd(&denom[(size_t)d * 8 + h], ex);
    }
}

// one thread per (edge, head): recompute alpha, scatter 8 channels
__global__ void gat_edge_msg1(const long long* __restrict__ ei, int E, int Nn,
                              const float* __restrict__ asrc, const float* __restrict__ adst,
                              const unsigned* __restrict__ emax, const float* __restrict__ denom,
                              const float* __restrict__ h1, float* __restrict__ out1) {
    long long tid = (long long)blockIdx.x * blockDim.x + threadIdx.x;
    long long tot = (long long)(E + Nn) * 8;
    if (tid >= tot) return;
    int e = (int)(tid >> 3);
    int h = (int)(tid & 7);
    int s, d; load_edge(ei, e, E, Nn, s, d);
    float v = asrc[(size_t)s * 8 + h] + adst[(size_t)d * 8 + h];
    v = v >= 0.f ? v : NEG_SLOPE * v;
    float alpha = __expf(v - ord2f(emax[(size_t)d * 8 + h])) /
                  (denom[(size_t)d * 8 + h] + 1e-16f);
    const float* hs = h1 + (size_t)s * 64 + h * 8;
    float* od = out1 + (size_t)d * 64 + h * 8;
#pragma unroll
    for (int c = 0; c < 8; ++c) atomicAdd(&od[c], hs[c] * alpha);
}

// =====================  bias + ELU  =====================
__global__ void gat_elu_bias1(float* __restrict__ out1, const float* __restrict__ b1, int Nn) {
    int i = blockIdx.x * blockDim.x + threadIdx.x;
    size_t tot = (size_t)Nn * 64;
    if ((size_t)i >= tot) return;
    float v = out1[i] + b1[i & 63];
    out1[i] = v > 0.f ? v : (__expf(v) - 1.f);
}

// =====================  layer-2 edge passes (H=1, C=40)  =====================
__global__ void gat_edge_max2(const long long* __restrict__ ei, int E, int Nn,
                              const float* __restrict__ asrc, const float* __restrict__ adst,
                              unsigned* __restrict__ emax) {
    int e = blockIdx.x * blockDim.x + threadIdx.x;
    if (e >= E + Nn) return;
    int s, d; load_edge(ei, e, E, Nn, s, d);
    float v = asrc[s] + adst[d];
    v = v >= 0.f ? v : NEG_SLOPE * v;
    atomicMax(&emax[d], f2ord(v));
}

__global__ void gat_edge_sum2(const long long* __restrict__ ei, int E, int Nn,
                              const float* __restrict__ asrc, const float* __restrict__ adst,
                              const unsigned* __restrict__ emax, float* __restrict__ denom) {
    int e = blockIdx.x * blockDim.x + threadIdx.x;
    if (e >= E + Nn) return;
    int s, d; load_edge(ei, e, E, Nn, s, d);
    float v = asrc[s] + adst[d];
    v = v >= 0.f ? v : NEG_SLOPE * v;
    atomicAdd(&denom[d], __expf(v - ord2f(emax[d])));
}

// one thread per (edge, 8-channel slab), 5 slabs cover 40 channels
__global__ void gat_edge_msg2(const long long* __restrict__ ei, int E, int Nn,
                              const float* __restrict__ asrc, const float* __restrict__ adst,
                              const unsigned* __restrict__ emax, const float* __restrict__ denom,
                              const float* __restrict__ h2, float* __restrict__ out2) {
    long long tid = (long long)blockIdx.x * blockDim.x + threadIdx.x;
    long long tot = (long long)(E + Nn) * 5;
    if (tid >= tot) return;
    int e = (int)(tid / 5);
    int p = (int)(tid % 5);
    int s, d; load_edge(ei, e, E, Nn, s, d);
    float v = asrc[s] + adst[d];
    v = v >= 0.f ? v : NEG_SLOPE * v;
    float alpha = __expf(v - ord2f(emax[d])) / (denom[d] + 1e-16f);
    const float* hs = h2 + (size_t)s * 40 + p * 8;
    float* od = out2 + (size_t)d * 40 + p * 8;
#pragma unroll
    for (int c = 0; c < 8; ++c) atomicAdd(&od[c], hs[c] * alpha);
}

// =====================  bias + log_softmax (in place)  =====================
__global__ void gat_final(float* __restrict__ out, const float* __restrict__ b2, int Nn) {
    int n = blockIdx.x * blockDim.x + threadIdx.x;
    if (n >= Nn) return;
    float v[40];
    float m = -3.4e38f;
#pragma unroll
    for (int c = 0; c < 40; ++c) {
        v[c] = out[(size_t)n * 40 + c] + b2[c];
        m = fmaxf(m, v[c]);
    }
    float ssum = 0.f;
#pragma unroll
    for (int c = 0; c < 40; ++c) ssum += __expf(v[c] - m);
    float lse = m + __logf(ssum);
#pragma unroll
    for (int c = 0; c < 40; ++c) out[(size_t)n * 40 + c] = v[c] - lse;
}

extern "C" void kernel_launch(void* const* d_in, const int* in_sizes, int n_in,
                              void* d_out, int out_size, void* d_ws, size_t ws_size,
                              hipStream_t stream) {
    const float*     x    = (const float*)d_in[0];
    const long long* ei   = (const long long*)d_in[1];   // int64 [2,E]
    const float*     W1   = (const float*)d_in[2];
    const float*     as1  = (const float*)d_in[3];
    const float*     ad1  = (const float*)d_in[4];
    const float*     b1   = (const float*)d_in[5];
    const float*     W2   = (const float*)d_in[6];
    const float*     as2  = (const float*)d_in[7];
    const float*     ad2  = (const float*)d_in[8];
    const float*     b2   = (const float*)d_in[9];
    float* dout = (float*)d_out;

    const int N = in_sizes[0] / 128;
    const int E = in_sizes[1] / 2;

    // workspace carve-up (floats)
    float* h1    = (float*)d_ws;                 // N*64   (reused as h2 later)
    float* out1  = h1   + (size_t)N * 64;        // N*64
    float* asr1  = out1 + (size_t)N * 64;        // N*8
    float* adt1  = asr1 + (size_t)N * 8;         // N*8
    unsigned* em1 = (unsigned*)(adt1 + (size_t)N * 8); // N*8
    float* dn1   = (float*)(em1 + (size_t)N * 8);      // N*8
    float* asr2  = dn1  + (size_t)N * 8;         // N
    float* adt2  = asr2 + N;                     // N
    unsigned* em2 = (unsigned*)(adt2 + N);       // N
    float* dn2   = (float*)(em2 + N);            // N
    float* h2    = h1;                           // reuse after layer-1 scatter

    const int BLK = 256;
    int mtiles = (N + 15) / 16;
    int gemmBlocks = (mtiles + 7) / 8;           // 8 waves / block
    int tot = E + N;

    // ---- layer 1 ----
    gat_gemm1_wmma<<<gemmBlocks, BLK, 0, stream>>>(x, W1, h1, N);
    gat_init1<<<((size_t)N * 64 + BLK - 1) / BLK, BLK, 0, stream>>>(out1, em1, dn1, N);
    gat_att1<<<(N + BLK - 1) / BLK, BLK, 0, stream>>>(h1, as1, ad1, asr1, adt1, N);
    gat_edge_max1<<<(tot + BLK - 1) / BLK, BLK, 0, stream>>>(ei, E, N, asr1, adt1, em1);
    gat_edge_sum1<<<(tot + BLK - 1) / BLK, BLK, 0, stream>>>(ei, E, N, asr1, adt1, em1, dn1);
    {
        long long t = (long long)tot * 8;
        gat_edge_msg1<<<(unsigned)((t + BLK - 1) / BLK), BLK, 0, stream>>>(
            ei, E, N, asr1, adt1, em1, dn1, h1, out1);
    }
    gat_elu_bias1<<<((size_t)N * 64 + BLK - 1) / BLK, BLK, 0, stream>>>(out1, b1, N);

    // ---- layer 2 ----
    gat_gemm2_wmma<<<gemmBlocks, BLK, 0, stream>>>(out1, W2, h2, N);
    gat_init2<<<((size_t)N * 40 + BLK - 1) / BLK, BLK, 0, stream>>>(dout, em2, dn2, N);
    gat_att2<<<(N + BLK - 1) / BLK, BLK, 0, stream>>>(h2, as2, ad2, asr2, adt2, N);
    gat_edge_max2<<<(tot + BLK - 1) / BLK, BLK, 0, stream>>>(ei, E, N, asr2, adt2, em2);
    gat_edge_sum2<<<(tot + BLK - 1) / BLK, BLK, 0, stream>>>(ei, E, N, asr2, adt2, em2, dn2);
    {
        long long t = (long long)tot * 5;
        gat_edge_msg2<<<(unsigned)((t + BLK - 1) / BLK), BLK, 0, stream>>>(
            ei, E, N, asr2, adt2, em2, dn2, h2, dout);
    }
    gat_final<<<(N + BLK - 1) / BLK, BLK, 0, stream>>>(dout, b2, N);
}
